// AMAdaptiveSelfAttention_45543833206869
// MI455X (gfx1250) — matrix operand order repaired and checked by reading, not compiled, not run on hardware
//
#include <hip/hip_runtime.h>
#include <hip/hip_bf16.h>

typedef __bf16 bf16_t;
typedef __attribute__((ext_vector_type(16))) __bf16 v16bf;
typedef __attribute__((ext_vector_type(8)))  __bf16 v8bf;
typedef __attribute__((ext_vector_type(8)))  float  v8f;

#define DIM 256
#define NTOK 4096
#define L2E 1.44269504088896340736f

// ---------------------------------------------------------------------------
// Kernel 1: convert h -> bf16, pack 5 weight matrices -> bf16, pack 4 biases.
// Wb rows: [0,256)=Wq [256,512)=Wk [512,768)=Wv [768,1024)=Wg [1024,1280)=Wo
// ---------------------------------------------------------------------------
__global__ void convert_pack(const float* __restrict__ h,
                             const float* __restrict__ Wq, const float* __restrict__ Wk,
                             const float* __restrict__ Wv, const float* __restrict__ Wg,
                             const float* __restrict__ Wo,
                             const float* __restrict__ bq, const float* __restrict__ bk,
                             const float* __restrict__ bv, const float* __restrict__ bg,
                             bf16_t* __restrict__ hb, bf16_t* __restrict__ Wb,
                             float* __restrict__ biasAll) {
    const int idx = blockIdx.x * blockDim.x + threadIdx.x;   // grid covers 4096*256 = N*D
    hb[idx] = (bf16_t)h[idx];
    if (idx < DIM * DIM) {
        Wb[idx]                 = (bf16_t)Wq[idx];
        Wb[1 * DIM * DIM + idx] = (bf16_t)Wk[idx];
        Wb[2 * DIM * DIM + idx] = (bf16_t)Wv[idx];
        Wb[3 * DIM * DIM + idx] = (bf16_t)Wg[idx];
        Wb[4 * DIM * DIM + idx] = (bf16_t)Wo[idx];
    }
    if (idx < DIM) {
        biasAll[idx]           = bq[idx];
        biasAll[DIM + idx]     = bk[idx];
        biasAll[2 * DIM + idx] = bv[idx];
        biasAll[3 * DIM + idx] = bg[idx];
    }
}

// ---------------------------------------------------------------------------
// Kernel 2/4: C[M x outCols] = A[M x 256](bf16) @ W[outCols x 256]^T (bf16) + bias
// Register-blocked 2x2: each wave owns a 32x32 output tile (4 f32 accumulators),
// reusing each A/B fragment twice -> 4 WMMAs per 8 b128 loads.
// Block = 8 waves as 4(M) x 2(N) -> 128x64 tile. WMMA layouts per ISA 7.12.2.
// ---------------------------------------------------------------------------
__global__ void wmma_gemm_bf16(const bf16_t* __restrict__ A, const bf16_t* __restrict__ W,
                               const float* __restrict__ bias, float* __restrict__ C,
                               int outCols) {
    const int lane  = threadIdx.x & 31;
    const int wave  = threadIdx.x >> 5;
    const int waveM = wave >> 1;          // 0..3
    const int waveN = wave & 1;           // 0..1
    const int m0    = blockIdx.x * 128 + waveM * 32;
    const int n0    = blockIdx.y * 64 + waveN * 32;
    const int half  = lane >> 4;          // 0: lanes 0-15, 1: lanes 16-31
    const int l16   = lane & 15;

    // A fragment: lane holds row; K-chunks half*8..+7 and +16..+23
    const bf16_t* arow0 = A + (size_t)(m0 + l16) * DIM + half * 8;
    const bf16_t* arow1 = arow0 + 16 * DIM;
    // B fragment: lane holds weight row (= output col); K-chunks half*8..+7 and +16..+23
    const bf16_t* brow0 = W + (size_t)(n0 + l16) * DIM + half * 8;
    const bf16_t* brow1 = brow0 + 16 * DIM;

    v8f c00 = {}, c01 = {}, c10 = {}, c11 = {};
    #pragma unroll
    for (int kk = 0; kk < DIM; kk += 32) {
        union Frag { v16bf v; struct { v8bf lo, hi; } p; } a0, a1, b0, b1;
        a0.p.lo = *(const v8bf*)(arow0 + kk);
        a0.p.hi = *(const v8bf*)(arow0 + kk + 16);
        a1.p.lo = *(const v8bf*)(arow1 + kk);
        a1.p.hi = *(const v8bf*)(arow1 + kk + 16);
        b0.p.lo = *(const v8bf*)(brow0 + kk);
        b0.p.hi = *(const v8bf*)(brow0 + kk + 16);
        b1.p.lo = *(const v8bf*)(brow1 + kk);
        b1.p.hi = *(const v8bf*)(brow1 + kk + 16);
        c00 = __builtin_amdgcn_wmma_f32_16x16x32_bf16(false, a0.v, false, b0.v,
                                                      (short)0, c00, false, false);
        c01 = __builtin_amdgcn_wmma_f32_16x16x32_bf16(false, a0.v, false, b1.v,
                                                      (short)0, c01, false, false);
        c10 = __builtin_amdgcn_wmma_f32_16x16x32_bf16(false, a1.v, false, b0.v,
                                                      (short)0, c10, false, false);
        c11 = __builtin_amdgcn_wmma_f32_16x16x32_bf16(false, a1.v, false, b1.v,
                                                      (short)0, c11, false, false);
    }

    // C/D layout: VGPR r, lane l -> M = r + (l>>4)*8, N = l&15
    const float bv0 = bias[n0 + l16];
    const float bv1 = bias[n0 + 16 + l16];
    float* cp0 = C + (size_t)(m0 + half * 8) * outCols + (n0 + l16);
    float* cp1 = cp0 + (size_t)16 * outCols;
    #pragma unroll
    for (int r = 0; r < 8; ++r) {
        cp0[(size_t)r * outCols]      = c00[r] + bv0;
        cp0[(size_t)r * outCols + 16] = c01[r] + bv1;
        cp1[(size_t)r * outCols]      = c10[r] + bv0;
        cp1[(size_t)r * outCols + 16] = c11[r] + bv1;
    }
}

// ---------------------------------------------------------------------------
// Kernel 3: per-token rank-1 attention.
// K/V staged into LDS via GLOBAL_LOAD_ASYNC_TO_LDS_B32 (ASYNCcnt-tracked,
// memory->LDS without VGPR bounce). Row max = max(a*kmax, a*kmin) via one
// block reduce. Inner loop: fma + v_exp_f32 (TRANS) co-executing with VALU.
// ---------------------------------------------------------------------------
__global__ void attention_kernel(const float* __restrict__ qkvg, bf16_t* __restrict__ ctxb) {
    __shared__ __align__(16) float ks[DIM];
    __shared__ __align__(16) float vs[DIM];
    __shared__ float rmax[DIM];
    __shared__ float rmin[DIM];

    const int n = blockIdx.x;
    const int i = threadIdx.x;
    const float* base = qkvg + (size_t)n * 1024;

    // Async global -> LDS staging of k and v (one b32 per lane each).
    {
        const unsigned long long gaK = (unsigned long long)(base + DIM + i);
        const unsigned long long gaV = (unsigned long long)(base + 2 * DIM + i);
        const unsigned ldsK = (unsigned)(unsigned long long)(void*)&ks[i];
        const unsigned ldsV = (unsigned)(unsigned long long)(void*)&vs[i];
        asm volatile("global_load_async_to_lds_b32 %0, %1, off"
                     :: "v"(ldsK), "v"(gaK) : "memory");
        asm volatile("global_load_async_to_lds_b32 %0, %1, off"
                     :: "v"(ldsV), "v"(gaV) : "memory");
        asm volatile("s_wait_asynccnt 0" ::: "memory");
    }
    __syncthreads();

    const float kv = ks[i];
    rmax[i] = kv;
    rmin[i] = kv;
    __syncthreads();

    #pragma unroll
    for (int s = 128; s > 0; s >>= 1) {
        if (i < s) {
            rmax[i] = fmaxf(rmax[i], rmax[i + s]);
            rmin[i] = fminf(rmin[i], rmin[i + s]);
        }
        __syncthreads();
    }
    const float kmax = rmax[0], kmin = rmin[0];

    const float a  = base[i] * 0.0625f;               // q_i / sqrt(256)
    const float m  = fmaxf(a * kmax, a * kmin);       // exact row max
    const float al = a * L2E;
    const float ml = m * L2E;

    float ssum = 0.f, acc = 0.f;
    const float4* k4 = (const float4*)ks;
    const float4* v4 = (const float4*)vs;
    #pragma unroll 4
    for (int j = 0; j < DIM / 4; ++j) {
        const float4 kk = k4[j];
        const float4 vv = v4[j];
        const float t0 = __builtin_amdgcn_exp2f(fmaf(al, kk.x, -ml));
        const float t1 = __builtin_amdgcn_exp2f(fmaf(al, kk.y, -ml));
        const float t2 = __builtin_amdgcn_exp2f(fmaf(al, kk.z, -ml));
        const float t3 = __builtin_amdgcn_exp2f(fmaf(al, kk.w, -ml));
        ssum += (t0 + t1) + (t2 + t3);
        acc = fmaf(t0, vv.x, acc);
        acc = fmaf(t1, vv.y, acc);
        acc = fmaf(t2, vv.z, acc);
        acc = fmaf(t3, vv.w, acc);
    }
    ctxb[(size_t)n * DIM + i] = (bf16_t)(acc / ssum);
}

// ---------------------------------------------------------------------------
// Kernel 5: fused = h + sigmoid(g) * o ; LayerNorm(fused) * gamma + beta
// ---------------------------------------------------------------------------
__global__ void gate_resid_ln(const float* __restrict__ h, const float* __restrict__ qkvg,
                              const float* __restrict__ o, const float* __restrict__ gamma,
                              const float* __restrict__ beta, float* __restrict__ out) {
    __shared__ float rs[DIM];
    __shared__ float rs2[DIM];
    const int n = blockIdx.x;
    const int i = threadIdx.x;

    const float hv  = h[(size_t)n * DIM + i];
    const float g   = qkvg[(size_t)n * 1024 + 3 * DIM + i];
    const float sig = 1.f / (1.f + __builtin_amdgcn_exp2f(-g * L2E));
    const float f   = fmaf(sig, o[(size_t)n * DIM + i], hv);

    rs[i]  = f;
    rs2[i] = f * f;
    __syncthreads();
    #pragma unroll
    for (int s = 128; s > 0; s >>= 1) {
        if (i < s) { rs[i] += rs[i + s]; rs2[i] += rs2[i + s]; }
        __syncthreads();
    }
    const float mu  = rs[0] * (1.f / DIM);
    const float var = rs2[0] * (1.f / DIM) - mu * mu;
    const float r   = __builtin_amdgcn_rsqf(var + 1e-5f);
    out[(size_t)n * DIM + i] = (f - mu) * r * gamma[i] + beta[i];
}

// ---------------------------------------------------------------------------
// Workspace layout (bytes):
//   hb      @ 0         : 4096*256 bf16    =  2,097,152
//   Wb      @ 2097152   : 1280*256 bf16    =    655,360
//   biasAll @ 2752512   : 1024 f32         =      4,096
//   qkvg    @ 2756608   : 4096*1024 f32    = 16,777,216
//   ctxb    @ 19533824  : 4096*256 bf16    =  2,097,152
//   o       @ 21630976  : 4096*256 f32     =  4,194,304   (end ~25.8 MB)
// ---------------------------------------------------------------------------
extern "C" void kernel_launch(void* const* d_in, const int* in_sizes, int n_in,
                              void* d_out, int out_size, void* d_ws, size_t ws_size,
                              hipStream_t stream) {
    const float* h     = (const float*)d_in[0];
    const float* Wq    = (const float*)d_in[1];
    const float* bq    = (const float*)d_in[2];
    const float* Wk    = (const float*)d_in[3];
    const float* bk    = (const float*)d_in[4];
    const float* Wv    = (const float*)d_in[5];
    const float* bv    = (const float*)d_in[6];
    const float* Wo    = (const float*)d_in[7];
    const float* bo    = (const float*)d_in[8];
    const float* Wg    = (const float*)d_in[9];
    const float* bg    = (const float*)d_in[10];
    const float* gamma = (const float*)d_in[11];
    const float* beta  = (const float*)d_in[12];
    float* out = (float*)d_out;

    char* ws = (char*)d_ws;
    bf16_t* hb      = (bf16_t*)(ws + 0);
    bf16_t* Wb      = (bf16_t*)(ws + 2097152);
    float*  biasAll = (float*)(ws + 2752512);
    float*  qkvg    = (float*)(ws + 2756608);
    bf16_t* ctxb    = (bf16_t*)(ws + 19533824);
    float*  o       = (float*)(ws + 21630976);

    convert_pack<<<NTOK, DIM, 0, stream>>>(h, Wq, Wk, Wv, Wg, Wo, bq, bk, bv, bg,
                                           hb, Wb, biasAll);
    // Q|K|V|G fused GEMM: [4096 x 1024], 128x64 block tile
    wmma_gemm_bf16<<<dim3(NTOK / 128, 1024 / 64), 256, 0, stream>>>(hb, Wb, biasAll,
                                                                    qkvg, 1024);
    attention_kernel<<<NTOK, DIM, 0, stream>>>(qkvg, ctxb);
    // O-projection GEMM: [4096 x 256]
    wmma_gemm_bf16<<<dim3(NTOK / 128, DIM / 64), 256, 0, stream>>>(ctxb,
                                                                   Wb + 4 * DIM * DIM,
                                                                   bo, o, DIM);
    gate_resid_ln<<<NTOK, DIM, 0, stream>>>(h, qkvg, o, gamma, beta, out);
}
